// GraphNet_1Trunk_44684839747697
// MI455X (gfx1250) — compile-verified, matrix operand order checked by reading, not run on hardware
//
#include <hip/hip_runtime.h>
#include <hip/hip_bf16.h>
#include <math.h>

// ---------------------------------------------------------------------------
// Types for CDNA5 WMMA (wave32, 16x16x32 bf16 -> f32 accumulate)
// ---------------------------------------------------------------------------
typedef __attribute__((ext_vector_type(16))) __bf16 v16bf;
typedef __attribute__((ext_vector_type(8)))  __bf16 v8bf;
typedef __attribute__((ext_vector_type(4)))  __bf16 v4bf;
typedef __attribute__((ext_vector_type(8)))  float  v8f;

#define N_NODES 16384
#define C_IN    128
#define H1      256
#define H2D     512
#define G_GRP   256
#define A_ACT   64
#define RH      4
#define DH      128

// ---------------------------------------------------------------------------
// Fragment loader. For the CDNA5 16-bit 16x32 A/B layouts (ISA 7.12.2), a
// lane's 16 elements are two contiguous 8-element (16B) runs at p and p+16:
//   lane<16: K base 0, lane>=16: K base 8; elems {base..base+7, base+16..+23}.
// Works for A rows (global bf16, row-major over K) and B cols (LDS, ld=64).
// ---------------------------------------------------------------------------
static __device__ __forceinline__ v16bf load_frag(const __bf16* p) {
  v8bf lo = *(const v8bf*)p;
  v8bf hi = *(const v8bf*)(p + 16);
  v16bf f;
#pragma unroll
  for (int i = 0; i < 8; ++i) { f[i] = lo[i]; f[8 + i] = hi[i]; }
  return f;
}

// ---------------------------------------------------------------------------
// GEMM: C[M,N] = A[M,K](bf16) * Wt[N,K](bf16) (+bias), f32 out.
// Block = 128 thr (4 waves). Wave w: rows blockIdx.x*64+16w, cols blockIdx.y*64..+64.
// B tile (64 cols x 64 K, 8KB) staged in LDS and shared by the 4 waves;
// two K-substeps per stage -> 8 WMMAs per iteration.
// ---------------------------------------------------------------------------
__global__ void gn_gemm_wmma(const __bf16* __restrict__ A,
                             const __bf16* __restrict__ Wt,
                             const float* __restrict__ bias,
                             float* __restrict__ C,
                             int M, int K, int Nn) {
  __shared__ __bf16 Bs[64 * 64];
  const int lane = threadIdx.x & 31;
  const int wave = threadIdx.x >> 5;
  const int row0 = blockIdx.x * 64 + wave * 16;
  const int col0 = blockIdx.y * 64;

  const int bcol = threadIdx.x >> 1;           // 0..63 : staging column
  const int bq   = (threadIdx.x & 1) * 32;     // 0/32  : staging K-half
  const int kb   = (lane < 16) ? 0 : 8;        // fragment K base
  const __bf16* arow = A + (size_t)(row0 + (lane & 15)) * K;

  v8f acc[4] = {v8f{}, v8f{}, v8f{}, v8f{}};
  for (int k0 = 0; k0 < K; k0 += 64) {
    // ---- stage B tile: each thread copies 64B (4 x b128) ----
    const float4* s4 = (const float4*)(Wt + (size_t)(col0 + bcol) * K + k0 + bq);
    float4* d4 = (float4*)(Bs + bcol * 64 + bq);
    d4[0] = s4[0]; d4[1] = s4[1]; d4[2] = s4[2]; d4[3] = s4[3];
    if (k0 + 64 < K) __builtin_prefetch((const char*)s4 + 128, 0, 1); // next K chunk
    __syncthreads();
#pragma unroll
    for (int ks = 0; ks < 64; ks += 32) {
      v16bf a = load_frag(arow + k0 + ks + kb);
#pragma unroll
      for (int j = 0; j < 4; ++j) {
        v16bf b = load_frag(Bs + (16 * j + (lane & 15)) * 64 + ks + kb);
        acc[j] = __builtin_amdgcn_wmma_f32_16x16x32_bf16(false, a, false, b,
                                                         (short)0, acc[j], false, false);
      }
    }
    __syncthreads();
  }
  // C/D layout: vgpr r, lane<16 -> M=r, N=lane ; lane>=16 -> M=8+r, N=lane-16
  const int crow = row0 + ((lane < 16) ? 0 : 8);
  const int ccol = col0 + (lane & 15);
#pragma unroll
  for (int j = 0; j < 4; ++j) {
    float bv = bias ? bias[ccol + 16 * j] : 0.0f;
#pragma unroll
    for (int r = 0; r < 8; ++r) {
      C[(size_t)(crow + r) * Nn + ccol + 16 * j] = acc[j][r] + bv;
    }
  }
}

// ---------------------------------------------------------------------------
// Weight prep: Wt[n*K+k] = (bf16) W[k*Nn+n]   (transpose+cast)
//              Wt[i]     = (bf16) W[i]        (cast copy, W already [n][k])
// Activation cast: f32 -> bf16, 4 elements/thread (b128 load, b64 store).
// ---------------------------------------------------------------------------
__global__ void gn_cast_transpose(const float* __restrict__ W, __bf16* __restrict__ Wt,
                                  int K, int Nn) {
  int idx = blockIdx.x * blockDim.x + threadIdx.x;
  if (idx >= K * Nn) return;
  int n = idx / K, k = idx - n * K;
  Wt[idx] = (__bf16)W[(size_t)k * Nn + n];
}
__global__ void gn_cast_copy(const float* __restrict__ W, __bf16* __restrict__ Wt, int count) {
  int idx = blockIdx.x * blockDim.x + threadIdx.x;
  if (idx < count) Wt[idx] = (__bf16)W[idx];
}
__global__ void gn_cast4(const float* __restrict__ X, __bf16* __restrict__ Y, int count4) {
  int i = blockIdx.x * blockDim.x + threadIdx.x;
  if (i >= count4) return;
  float4 v = ((const float4*)X)[i];
  v4bf o;
  o[0] = (__bf16)v.x; o[1] = (__bf16)v.y; o[2] = (__bf16)v.z; o[3] = (__bf16)v.w;
  ((v4bf*)Y)[i] = o;
}

// ---------------------------------------------------------------------------
// Per-node dots: hs[n] = h[n].a_src, hd[n] = h[n].a_dst   (wave per node)
// ---------------------------------------------------------------------------
__global__ void gn_node_dots(const float* __restrict__ Hm,
                             const float* __restrict__ asrc,
                             const float* __restrict__ adst,
                             float* __restrict__ hs, float* __restrict__ hd,
                             int N_, int H) {
  int lane = threadIdx.x & 31;
  int node = blockIdx.x * (blockDim.x >> 5) + (threadIdx.x >> 5);
  if (node >= N_) return;
  float a = 0.f, d = 0.f;
  for (int c = lane; c < H; c += 32) {
    float v = Hm[(size_t)node * H + c];
    a += v * asrc[c];
    d += v * adst[c];
  }
#pragma unroll
  for (int o = 16; o > 0; o >>= 1) { a += __shfl_xor(a, o, 32); d += __shfl_xor(d, o, 32); }
  if (lane == 0) { hs[node] = a; hd[node] = d; }
}

// ---------------------------------------------------------------------------
// Edge kernels. Self-loops are edges [E, E+N) with src=dst=e-E.
// Orderable-uint key for float atomicMax.
// ---------------------------------------------------------------------------
static __device__ __forceinline__ unsigned gn_fkey(float f) {
  unsigned u = __float_as_uint(f);
  return (u & 0x80000000u) ? ~u : (u | 0x80000000u);
}
static __device__ __forceinline__ float gn_funkey(unsigned u) {
  return __uint_as_float((u & 0x80000000u) ? (u & 0x7fffffffu) : ~u);
}

__global__ void gn_edge_max(const int* __restrict__ src, const int* __restrict__ dst,
                            int E_, int total,
                            const float* __restrict__ hs, const float* __restrict__ hd,
                            unsigned* __restrict__ mkey) {
  int e = blockIdx.x * blockDim.x + threadIdx.x;
  if (e >= total) return;
  int s = (e < E_) ? src[e] : (e - E_);
  int d = (e < E_) ? dst[e] : (e - E_);
  float v = hs[s] + hd[d];
  v = (v > 0.f) ? v : 0.2f * v;   // leaky_relu 0.2
  atomicMax(&mkey[d], gn_fkey(v));
}

__global__ void gn_edge_exp(const int* __restrict__ src, const int* __restrict__ dst,
                            int E_, int total,
                            const float* __restrict__ hs, const float* __restrict__ hd,
                            const unsigned* __restrict__ mkey,
                            float* __restrict__ ex, float* __restrict__ den) {
  int e = blockIdx.x * blockDim.x + threadIdx.x;
  if (e >= total) return;
  int s = (e < E_) ? src[e] : (e - E_);
  int d = (e < E_) ? dst[e] : (e - E_);
  float v = hs[s] + hd[d];
  v = (v > 0.f) ? v : 0.2f * v;
  float x = __expf(v - gn_funkey(mkey[d]));
  ex[e] = x;
  atomicAdd(&den[d], x);
}

__global__ void gn_edge_agg(const int* __restrict__ src, const int* __restrict__ dst,
                            int E_, int total,
                            const float* __restrict__ ex, const float* __restrict__ den,
                            const float* __restrict__ Hm, float* __restrict__ out, int H) {
  long long idx = (long long)blockIdx.x * blockDim.x + threadIdx.x;
  int cpe = H >> 2;
  int e  = (int)(idx / cpe);
  int c4 = (int)(idx % cpe) * 4;
  if (e >= total) return;
  int s = (e < E_) ? src[e] : (e - E_);
  int d = (e < E_) ? dst[e] : (e - E_);
  float alpha = ex[e] / den[d];
  float4 hv = *(const float4*)(Hm + (size_t)s * H + c4);
  float* o = out + (size_t)d * H + c4;
  atomicAdd(o + 0, alpha * hv.x);
  atomicAdd(o + 1, alpha * hv.y);
  atomicAdd(o + 2, alpha * hv.z);
  atomicAdd(o + 3, alpha * hv.w);
}

// ---------------------------------------------------------------------------
// BatchNorm (over N rows, per channel) fused with activation / residual.
// act: 0 = ELU, 1 = ReLU. If resid != null: out = relu(act(y) + resid).
// Block per channel, 256 threads.
// ---------------------------------------------------------------------------
__global__ void gn_bn_act(const float* __restrict__ X,
                          const float* __restrict__ g, const float* __restrict__ b,
                          const float* __restrict__ resid,
                          float* __restrict__ Y, int N_, int H, int act) {
  int c = blockIdx.x;
  __shared__ float s1[256], s2[256];
  float sum = 0.f, sq = 0.f;
  for (int r = threadIdx.x; r < N_; r += blockDim.x) {
    float v = X[(size_t)r * H + c];
    sum += v; sq += v * v;
  }
  s1[threadIdx.x] = sum; s2[threadIdx.x] = sq;
  __syncthreads();
  for (int s = 128; s > 0; s >>= 1) {
    if (threadIdx.x < (unsigned)s) { s1[threadIdx.x] += s1[threadIdx.x + s]; s2[threadIdx.x] += s2[threadIdx.x + s]; }
    __syncthreads();
  }
  float mu  = s1[0] / (float)N_;
  float var = s2[0] / (float)N_ - mu * mu;
  float scale = g[c] * rsqrtf(var + 1e-5f);
  float shift = b[c];
  for (int r = threadIdx.x; r < N_; r += blockDim.x) {
    float y = (X[(size_t)r * H + c] - mu) * scale + shift;
    float h = (act == 0) ? ((y > 0.f) ? y : (__expf(y) - 1.f)) : fmaxf(y, 0.f);
    if (resid) h = fmaxf(h + resid[(size_t)r * H + c], 0.f);
    Y[(size_t)r * H + c] = h;
  }
}

// ---------------------------------------------------------------------------
// Row dot: out[n] = dot(X[n,:], w) + b0  (optional tanh). Wave per row.
// ---------------------------------------------------------------------------
__global__ void gn_rowdot(const float* __restrict__ X, const float* __restrict__ w,
                          const float* __restrict__ b, float* __restrict__ out,
                          int rows, int H, int do_tanh) {
  int lane = threadIdx.x & 31;
  int r = blockIdx.x * (blockDim.x >> 5) + (threadIdx.x >> 5);
  if (r >= rows) return;
  float acc = 0.f;
  for (int c = lane; c < H; c += 32) acc += X[(size_t)r * H + c] * w[c];
#pragma unroll
  for (int o = 16; o > 0; o >>= 1) acc += __shfl_xor(acc, o, 32);
  if (lane == 0) {
    float v = acc + (b ? b[0] : 0.f);
    out[r] = do_tanh ? tanhf(v) : v;
  }
}

// Policy scatter: logits[gid*A + aid] += p[n]
__global__ void gn_scatter_p(const float* __restrict__ p, const int* __restrict__ batch,
                             float* __restrict__ logits, int N_) {
  int n = blockIdx.x * blockDim.x + threadIdx.x;
  if (n >= N_) return;
  int gid = batch[n * 3 + 0];
  int aid = batch[n * 3 + 1];
  atomicAdd(&logits[gid * A_ACT + aid], p[n]);
}

// log_softmax per group over A=64 actions. Block per group, 64 threads.
__global__ void gn_logsoftmax(const float* __restrict__ logits, float* __restrict__ out) {
  int g = blockIdx.x, t = threadIdx.x;
  __shared__ float sh[64];
  float v = logits[g * A_ACT + t];
  sh[t] = v; __syncthreads();
  for (int s = 32; s > 0; s >>= 1) { if (t < s) sh[t] = fmaxf(sh[t], sh[t + s]); __syncthreads(); }
  float m = sh[0]; __syncthreads();
  float ex = __expf(v - m);
  sh[t] = ex; __syncthreads();
  for (int s = 32; s > 0; s >>= 1) { if (t < s) sh[t] += sh[t + s]; __syncthreads(); }
  out[g * A_ACT + t] = v - (m + __logf(sh[0]));
}

// q[j] = sum_i Wq[j,i] + bq[j]  (wave per row j of in_w[0:512,:])
__global__ void gn_qvec(const float* __restrict__ in_w, const float* __restrict__ in_b,
                        float* __restrict__ q) {
  int lane = threadIdx.x & 31;
  int j = blockIdx.x * (blockDim.x >> 5) + (threadIdx.x >> 5);
  if (j >= H2D) return;
  float acc = 0.f;
  for (int i = lane; i < H2D; i += 32) acc += in_w[(size_t)j * H2D + i];
#pragma unroll
  for (int o = 16; o > 0; o >>= 1) acc += __shfl_xor(acc, o, 32);
  if (lane == 0) q[j] = acc + in_b[j];
}

// s[n,h] = dot(q[h,:], k[n,h,:]) / sqrt(Dh). Wave per (n,h).
__global__ void gn_attn_score(const float* __restrict__ kmat, const float* __restrict__ q,
                              float* __restrict__ s, int N_) {
  int lane = threadIdx.x & 31;
  int task = blockIdx.x * (blockDim.x >> 5) + (threadIdx.x >> 5);
  if (task >= N_ * RH) return;
  int n = task >> 2, h = task & 3;
  float acc = 0.f;
  for (int d = lane; d < DH; d += 32) acc += q[h * DH + d] * kmat[(size_t)n * H2D + h * DH + d];
#pragma unroll
  for (int o = 16; o > 0; o >>= 1) acc += __shfl_xor(acc, o, 32);
  if (lane == 0) s[n * RH + h] = acc * 0.088388347648f; // 1/sqrt(128)
}

// Group softmax over the 64 contiguous nodes of each group, per head.
// Block per group: 256 threads = 64 nodes x 4 heads. s -> alpha in place.
__global__ void gn_attn_softmax(float* __restrict__ s) {
  int g = blockIdx.x, t = threadIdx.x;
  int h = t & 3, i = t >> 2;
  int n = g * 64 + i;
  float v = s[n * RH + h];
  __shared__ float sh[256];
  sh[t] = v; __syncthreads();
  for (int st = 32; st > 0; st >>= 1) { if (i < st) sh[t] = fmaxf(sh[t], sh[t + st * 4]); __syncthreads(); }
  float m = sh[h]; __syncthreads();
  float ex = __expf(v - m);
  sh[t] = ex; __syncthreads();
  for (int st = 32; st > 0; st >>= 1) { if (i < st) sh[t] += sh[t + st * 4]; __syncthreads(); }
  float den = sh[h];
  s[n * RH + h] = ex / den;
}

// att[g, c] = sum_{i<64} alpha[g*64+i, c/128] * val[g*64+i, c]. Block per group.
__global__ void gn_attn_agg(const float* __restrict__ alpha, const float* __restrict__ val,
                            float* __restrict__ att) {
  int g = blockIdx.x;
  for (int c = threadIdx.x; c < H2D; c += blockDim.x) {
    int h = c >> 7;
    float acc = 0.f;
#pragma unroll 4
    for (int i = 0; i < 64; ++i) {
      int n = g * 64 + i;
      acc += alpha[n * RH + h] * val[(size_t)n * H2D + c];
    }
    att[(size_t)g * H2D + c] = acc;
  }
}

// ---------------------------------------------------------------------------
// Host orchestration
// ---------------------------------------------------------------------------
extern "C" void kernel_launch(void* const* d_in, const int* in_sizes, int n_in,
                              void* d_out, int out_size, void* d_ws, size_t ws_size,
                              hipStream_t stream) {
  const int E   = in_sizes[0] / 2;
  const int TOT = E + N_NODES;

  const int*   edge  = (const int*)d_in[0];
  const int*   srcp  = edge;
  const int*   dstp  = edge + E;
  const float* xin   = (const float*)d_in[1];
  const int*   batch = (const int*)d_in[2];

  // ---- workspace layout ----
  char* base = (char*)d_ws;
  size_t off = 0;
  auto alloc_f = [&](size_t n) { float* p = (float*)(base + off); off += n * sizeof(float); return p; };
  float* bufX = alloc_f((size_t)N_NODES * H2D);
  float* bufH = alloc_f((size_t)N_NODES * H2D);
  float* bufO = alloc_f((size_t)N_NODES * H2D);
  float* hs   = alloc_f(N_NODES);
  float* hd   = alloc_f(N_NODES);
  float* den  = alloc_f(N_NODES);
  float* pscal= alloc_f(N_NODES);
  unsigned* mkey = (unsigned*)(base + off); off += (size_t)N_NODES * 4;
  float* ex   = alloc_f(TOT);
  float* sbuf = alloc_f((size_t)N_NODES * RH);
  float* qv   = alloc_f(H2D);
  float* logits = alloc_f((size_t)G_GRP * A_ACT);
  float* att  = alloc_f((size_t)G_GRP * H2D);
  float* att2 = alloc_f((size_t)G_GRP * H2D);
  __bf16* wt   = (__bf16*)(base + off); off += (size_t)H2D * H2D * sizeof(__bf16);
  __bf16* abuf = (__bf16*)(base + off); off += (size_t)N_NODES * H2D * sizeof(__bf16);
  (void)ws_size; (void)n_in; (void)out_size;

  const int eb = (TOT + 255) / 256;

  auto cast_act = [&](const float* x, int count) {
    gn_cast4<<<(count / 4 + 255) / 256, 256, 0, stream>>>(x, abuf, count / 4);
  };

  // GAT layer: x[Kin] --W--> h[Kout] in hbuf, aggregate into obuf (no bias: BN absorbs it)
  auto run_gat = [&](const float* x, int Kin, int Kout, int pi, float* hbuf, float* obuf) {
    const float* W    = (const float*)d_in[pi];
    const float* asrc = (const float*)d_in[pi + 1];
    const float* adst = (const float*)d_in[pi + 2];
    gn_cast_transpose<<<(Kin * Kout + 255) / 256, 256, 0, stream>>>(W, wt, Kin, Kout);
    cast_act(x, N_NODES * Kin);
    gn_gemm_wmma<<<dim3(N_NODES / 64, Kout / 64), 128, 0, stream>>>(abuf, wt, nullptr, hbuf, N_NODES, Kin, Kout);
    gn_node_dots<<<N_NODES / 8, 256, 0, stream>>>(hbuf, asrc, adst, hs, hd, N_NODES, Kout);
    hipMemsetAsync(mkey, 0, (size_t)N_NODES * 4, stream);      // key 0 == -inf
    hipMemsetAsync(den, 0, (size_t)N_NODES * 4, stream);
    gn_edge_max<<<eb, 256, 0, stream>>>(srcp, dstp, E, TOT, hs, hd, mkey);
    gn_edge_exp<<<eb, 256, 0, stream>>>(srcp, dstp, E, TOT, hs, hd, mkey, ex, den);
    hipMemsetAsync(obuf, 0, (size_t)N_NODES * Kout * sizeof(float), stream);
    long long nthr = (long long)TOT * (Kout / 4);
    gn_edge_agg<<<(unsigned)((nthr + 255) / 256), 256, 0, stream>>>(srcp, dstp, E, TOT, ex, den, hbuf, obuf, Kout);
  };

  // -------- trunk --------
  run_gat(xin, C_IN, H1, 3, bufH, bufO);                       // gat1 (128->256)
  gn_bn_act<<<H1, 256, 0, stream>>>(bufO, (const float*)d_in[7], (const float*)d_in[8],
                                    nullptr, bufX, N_NODES, H1, 0);        // bn1 + elu
  run_gat(bufX, H1, H2D, 9, bufH, bufO);                       // gat2 (256->512)
  gn_bn_act<<<H2D, 256, 0, stream>>>(bufO, (const float*)d_in[13], (const float*)d_in[14],
                                     nullptr, bufX, N_NODES, H2D, 1);      // bn2 + relu
  for (int blk = 0; blk < 4; ++blk) {
    int p = 15 + blk * 12;
    run_gat(bufX, H2D, H2D, p, bufH, bufO);                    // gc1
    gn_bn_act<<<H2D, 256, 0, stream>>>(bufO, (const float*)d_in[p + 4], (const float*)d_in[p + 5],
                                       nullptr, bufO, N_NODES, H2D, 0);    // bn_a + elu (in place)
    run_gat(bufO, H2D, H2D, p + 6, bufH, bufO);                // gc2 (agg overwrites bufO after gemm)
    gn_bn_act<<<H2D, 256, 0, stream>>>(bufO, (const float*)d_in[p + 10], (const float*)d_in[p + 11],
                                       bufX, bufX, N_NODES, H2D, 1);       // bn_b + relu + resid
  }

  // -------- policy head --------
  cast_act(bufX, N_NODES * H2D);                               // trunk x -> bf16 (shared with v_lin)
  gn_cast_transpose<<<(H2D * H2D + 255) / 256, 256, 0, stream>>>((const float*)d_in[63], wt, H2D, H2D);
  gn_gemm_wmma<<<dim3(N_NODES / 64, H2D / 64), 128, 0, stream>>>(abuf, wt, nullptr, bufH, N_NODES, H2D, H2D);
  gn_bn_act<<<H2D, 256, 0, stream>>>(bufH, (const float*)d_in[65], (const float*)d_in[66],
                                     nullptr, bufH, N_NODES, H2D, 1);
  gn_rowdot<<<N_NODES / 8, 256, 0, stream>>>(bufH, (const float*)d_in[67], (const float*)d_in[68],
                                             pscal, N_NODES, H2D, 0);
  hipMemsetAsync(logits, 0, (size_t)G_GRP * A_ACT * sizeof(float), stream);
  gn_scatter_p<<<N_NODES / 256, 256, 0, stream>>>(pscal, batch, logits, N_NODES);
  gn_logsoftmax<<<G_GRP, 64, 0, stream>>>(logits, (float*)d_out);

  // -------- value head --------
  gn_cast_transpose<<<(H2D * H2D + 255) / 256, 256, 0, stream>>>((const float*)d_in[69], wt, H2D, H2D);
  gn_gemm_wmma<<<dim3(N_NODES / 64, H2D / 64), 128, 0, stream>>>(abuf, wt, nullptr, bufH, N_NODES, H2D, H2D);
  gn_bn_act<<<H2D, 256, 0, stream>>>(bufH, (const float*)d_in[71], (const float*)d_in[72],
                                     nullptr, bufH, N_NODES, H2D, 1);
  cast_act(bufH, N_NODES * H2D);
  gn_cast_transpose<<<(H2D * H2D + 255) / 256, 256, 0, stream>>>((const float*)d_in[73], wt, H2D, H2D);
  gn_gemm_wmma<<<dim3(N_NODES / 64, H2D / 64), 128, 0, stream>>>(abuf, wt, (const float*)d_in[74], bufO, N_NODES, H2D, H2D);

  const float* in_w = (const float*)d_in[75];
  const float* in_b = (const float*)d_in[76];
  gn_qvec<<<H2D / 8, 256, 0, stream>>>(in_w, in_b, qv);

  cast_act(bufO, N_NODES * H2D);                               // v2 -> bf16 (shared by k and val GEMMs)
  // k = v2 @ Wk^T + bk  (Wk rows of in_w are already [n][k] for our B loader)
  gn_cast_copy<<<(H2D * H2D + 255) / 256, 256, 0, stream>>>(in_w + (size_t)H2D * H2D, wt, H2D * H2D);
  gn_gemm_wmma<<<dim3(N_NODES / 64, H2D / 64), 128, 0, stream>>>(abuf, wt, in_b + H2D, bufH, N_NODES, H2D, H2D);
  gn_attn_score<<<(N_NODES * RH) / 8, 256, 0, stream>>>(bufH, qv, sbuf, N_NODES);
  gn_attn_softmax<<<G_GRP, 256, 0, stream>>>(sbuf);

  // val = v2 @ Wv^T + bv
  gn_cast_copy<<<(H2D * H2D + 255) / 256, 256, 0, stream>>>(in_w + (size_t)2 * H2D * H2D, wt, H2D * H2D);
  gn_gemm_wmma<<<dim3(N_NODES / 64, H2D / 64), 128, 0, stream>>>(abuf, wt, in_b + 2 * H2D, bufH, N_NODES, H2D, H2D);
  gn_attn_agg<<<G_GRP, 256, 0, stream>>>(sbuf, bufH, att);

  // att2 = att @ out_w^T + out_b
  cast_act(att, G_GRP * H2D);
  gn_cast_copy<<<(H2D * H2D + 255) / 256, 256, 0, stream>>>((const float*)d_in[77], wt, H2D * H2D);
  gn_gemm_wmma<<<dim3(G_GRP / 64, H2D / 64), 128, 0, stream>>>(abuf, wt, (const float*)d_in[78], att2, G_GRP, H2D, H2D);

  // vout = tanh(att2 @ v_read.W + b)
  gn_rowdot<<<G_GRP / 8, 256, 0, stream>>>(att2, (const float*)d_in[79], (const float*)d_in[80],
                                           (float*)d_out + G_GRP * A_ACT, G_GRP, H2D, 1);
}
